// SwiGLUTransformerEncoderLayer_86835648790980
// MI455X (gfx1250) — matrix operand order repaired and checked by reading, not compiled, not run on hardware
//
#include <hip/hip_runtime.h>
#include <hip/hip_bf16.h>

// ---------------------------------------------------------------------------
// Types for CDNA5 WMMA (wave32, 16x16x32 bf16 -> f32)
// ---------------------------------------------------------------------------
typedef __bf16        bf16x16 __attribute__((ext_vector_type(16)));
typedef float         f32x8   __attribute__((ext_vector_type(8)));
typedef unsigned int  u32x4   __attribute__((ext_vector_type(4)));
typedef int           i32x4v  __attribute__((ext_vector_type(4)));
typedef int           i32x8v  __attribute__((ext_vector_type(8)));

#ifndef __has_builtin
#define __has_builtin(x) 0
#endif
#if __has_builtin(__builtin_amdgcn_tensor_load_to_lds)
#define HAVE_TDM 1
#else
#define HAVE_TDM 0
#endif

union Frag {            // 32 bytes = 16 bf16 = one WMMA A or B operand (wave32)
    bf16x16 v;
    u32x4   q[2];
};

__device__ __forceinline__ f32x8 wmma_bf16(const Frag& a, const Frag& b, f32x8 c) {
    return __builtin_amdgcn_wmma_f32_16x16x32_bf16(false, a.v, false, b.v,
                                                   (short)0, c, false, false);
}

__device__ __forceinline__ unsigned short f32_bf16(float f) {   // RNE
    unsigned int u = __float_as_uint(f);
    unsigned int r = (u + 0x7FFFu + ((u >> 16) & 1u)) >> 16;
    return (unsigned short)r;
}
__device__ __forceinline__ unsigned int pack_bf16(float a, float b) {
    return (unsigned int)f32_bf16(a) | ((unsigned int)f32_bf16(b) << 16);
}

// ---------------------------------------------------------------------------
// TDM: 2-D tile (tile_dim0 = 64 bf16 elems = 128B rows) -> LDS, with
// pad_enable inserting 4 DWORDs (16B) after every 32 DWORDs (128B), i.e.
// LDS row stride = 72 elements.  D# layout per CDNA5 ISA ch.8.
// ---------------------------------------------------------------------------
#if HAVE_TDM
__device__ __forceinline__ void tdm_load_tile(unsigned lds_byte_off,
                                              const void* gaddr,
                                              unsigned tile_rows,
                                              unsigned tensor_d0,   // row len (elems)
                                              unsigned tensor_d1,   // total rows
                                              unsigned stride0) {   // row stride (elems)
    unsigned long long ga = (unsigned long long)(__SIZE_TYPE__)gaddr;
    u32x4 g0 = { 1u,                                  // count=1 (valid), user mode
                 lds_byte_off,                        // lds_addr
                 (unsigned)ga,                        // global_addr[31:0]
                 (unsigned)(ga >> 32) | (2u << 30) }; // global_addr[56:32] | type=2
    i32x8v g1;
    g1[0] = (int)((1u << 16)      // data_size = 1 -> 2 bytes
                | (1u << 20)      // pad_enable
                | (4u << 22)      // pad_interval: 2^(4+1)=32 DWORDs (=128B row)
                | (3u << 25));    // pad_amount: 3+1 = 4 DWORDs (=16B pad)
    g1[1] = (int)((tensor_d0 & 0xFFFFu) << 16);                  // dim0[15:0] @ bits63:48
    g1[2] = (int)((tensor_d0 >> 16) | ((tensor_d1 & 0xFFFFu) << 16));
    g1[3] = (int)((tensor_d1 >> 16) | (64u << 16));              // tile_dim0 = 64
    g1[4] = (int)(tile_rows & 0xFFFFu);                          // tile_dim1 (tile_dim2=0)
    g1[5] = (int)stride0;                                        // dim0_stride[31:0]
    g1[6] = 0;                                                   // dim0_stride[47:32]
    g1[7] = 0;
    i32x4v z4 = {0, 0, 0, 0};
#if __clang_major__ >= 23
    i32x8v z8 = {0, 0, 0, 0, 0, 0, 0, 0};
    __builtin_amdgcn_tensor_load_to_lds(g0, g1, z4, z4, z8, 0);
#else
    __builtin_amdgcn_tensor_load_to_lds(g0, g1, z4, z4, 0);
#endif
}
#endif

// ---------------------------------------------------------------------------
// Problem constants
// ---------------------------------------------------------------------------
#define BB   512
#define SS   65
#define DD   1024
#define HH   16
#define DSS  16
#define DHH  64
#define DFFD 4096
#define NTOK (BB * SS)          // 33280

// ---------------------------------------------------------------------------
// fp32 -> bf16 cast (weights)
// ---------------------------------------------------------------------------
__global__ void cast_bf16_kernel(const float* __restrict__ x,
                                 unsigned short* __restrict__ y, int n4) {
    int i = blockIdx.x * blockDim.x + threadIdx.x;
    if (i >= n4) return;
    float4 v = ((const float4*)x)[i];
    ((unsigned int*)y)[2 * i]     = pack_bf16(v.x, v.y);
    ((unsigned int*)y)[2 * i + 1] = pack_bf16(v.z, v.w);
}

// ---------------------------------------------------------------------------
// LayerNorm over D=1024 per token + bf16 cast (optionally also cast raw input)
// ---------------------------------------------------------------------------
__global__ __launch_bounds__(256)
void ln_cast_kernel(const float* __restrict__ x, const float* __restrict__ g,
                    const float* __restrict__ be,
                    unsigned short* __restrict__ ynorm,
                    unsigned short* __restrict__ ycast) {
    const int token = blockIdx.x;
    const int tid   = threadIdx.x;
    const float4 v  = ((const float4*)(x + (size_t)token * DD))[tid];

    float s  = v.x + v.y + v.z + v.w;
    float ss = v.x * v.x + v.y * v.y + v.z * v.z + v.w * v.w;
#pragma unroll
    for (int o = 16; o > 0; o >>= 1) {
        s  += __shfl_xor(s, o, 32);
        ss += __shfl_xor(ss, o, 32);
    }
    __shared__ float ls[8], lss[8];
    __shared__ float mu_s, rs_s;
    if ((tid & 31) == 0) { ls[tid >> 5] = s; lss[tid >> 5] = ss; }
    __syncthreads();
    if (tid == 0) {
        float S = 0.f, SSq = 0.f;
        for (int i = 0; i < 8; ++i) { S += ls[i]; SSq += lss[i]; }
        float mu  = S * (1.0f / DD);
        float var = SSq * (1.0f / DD) - mu * mu;
        mu_s = mu;
        rs_s = rsqrtf(var + 1e-5f);
    }
    __syncthreads();
    const float mu = mu_s, rs = rs_s;
    const float4 gv = ((const float4*)g)[tid];
    const float4 bv = ((const float4*)be)[tid];
    float n0 = (v.x - mu) * rs * gv.x + bv.x;
    float n1 = (v.y - mu) * rs * gv.y + bv.y;
    float n2 = (v.z - mu) * rs * gv.z + bv.z;
    float n3 = (v.w - mu) * rs * gv.w + bv.w;
    unsigned int* yp = (unsigned int*)(ynorm + (size_t)token * DD + tid * 4);
    yp[0] = pack_bf16(n0, n1);
    yp[1] = pack_bf16(n2, n3);
    if (ycast) {
        unsigned int* cp = (unsigned int*)(ycast + (size_t)token * DD + tid * 4);
        cp[0] = pack_bf16(v.x, v.y);
        cp[1] = pack_bf16(v.z, v.w);
    }
}

// ---------------------------------------------------------------------------
// WMMA GEMM with TDM->LDS double-buffered pipeline.
//   C[M,N] = A[M,K] @ W[N,K]^T (+epilogue)
//   EPI 0: store bf16               1: +bias[n], store bf16
//   EPI 2: +bias[n]+resid, f32      3: +resid, f32
//   EPI 4: SwiGLU dual-weight: silu(A@W^T) * (A@W2^T), store bf16 (W2=W+N*K)
// 128 threads = 4 waves (2x2); per-wave tile (BM/2)x(BN/2); K staged 64/step.
// LDS tile rows padded to 72 elems (TDM pad_enable) for bank spread.
// ---------------------------------------------------------------------------
template <int BM, int BN, int EPI>
__global__ __launch_bounds__(128)
void gemm_wmma_kernel(const unsigned short* __restrict__ A,
                      const unsigned short* __restrict__ W,
                      const float* __restrict__ bias,
                      const float* __restrict__ resid,
                      void* __restrict__ out, int M, int N, int K) {
    static_assert(BM == BN, "square block tiles");
    constexpr int WM = BM / 2, WN = BN / 2;
    constexpr int TM = WM / 16, TN = WN / 16;
    constexpr int LDK = 72;                        // 64 + 8 pad elements
    constexpr int NTILE = (EPI == 4) ? 3 : 2;      // A, W [, W2]

    __shared__ alignas(16) unsigned short Lds[2][NTILE][BM][LDK];

    const int lane = threadIdx.x & 31;
    const int wave = threadIdx.x >> 5;
    const int wmL  = (wave >> 1) * WM;             // block-local row
    const int wnL  = (wave & 1) * WN;              // block-local col
    const int wm   = blockIdx.y * BM + wmL;
    const int wn   = blockIdx.x * BN + wnL;
    const int r16  = lane & 15;
    const int aoff = (lane < 16) ? 0 : 8;          // A-frag K sub-offset
    const int boff = (lane < 16) ? 0 : 16;         // unused; B frags also use aoff
    const int mhi  = (lane < 16) ? 0 : 8;          // C-frag row offset

    const unsigned short* gA  = A + (size_t)blockIdx.y * BM * K;
    const unsigned short* gW  = W + (size_t)blockIdx.x * BN * K;
    const unsigned short* gW2 = W + (size_t)N * K + (size_t)blockIdx.x * BN * K;

    auto stage = [&](int buf, int k) {
#if HAVE_TDM
        if (wave == 0) {
            tdm_load_tile((unsigned)(__SIZE_TYPE__)&Lds[buf][0][0][0],
                          gA + k, BM, (unsigned)K, (unsigned)M, (unsigned)K);
            tdm_load_tile((unsigned)(__SIZE_TYPE__)&Lds[buf][1][0][0],
                          gW + k, BN, (unsigned)K, (unsigned)N, (unsigned)K);
            if constexpr (EPI == 4)
                tdm_load_tile((unsigned)(__SIZE_TYPE__)&Lds[buf][2][0][0],
                              gW2 + k, BN, (unsigned)K, (unsigned)N, (unsigned)K);
        }
#else   // cooperative-copy fallback (also what the host pass parses)
        for (int idx = threadIdx.x; idx < NTILE * BM * 8; idx += 128) {
            const int t = idx / (BM * 8);
            const int r = (idx >> 3) % BM;
            const int c = idx & 7;
            const unsigned short* g = (t == 0) ? gA : (t == 1) ? gW : gW2;
            *(u32x4*)&Lds[buf][t][r][c * 8] =
                *(const u32x4*)(g + (size_t)r * K + k + c * 8);
        }
#endif
    };

    f32x8 acc[TM][TN] = {};
    f32x8 acc2[EPI == 4 ? TM : 1][EPI == 4 ? TN : 1] = {};

    int buf = 0;
    stage(0, 0);
    for (int k = 0; k < K; k += 64) {
#if HAVE_TDM
        if (wave == 0) __builtin_amdgcn_s_wait_tensorcnt((short)0);
#endif
        __syncthreads();
        if (k + 64 < K) stage(buf ^ 1, k + 64);

        const unsigned short* Ab  = &Lds[buf][0][0][0];
        const unsigned short* Wb  = &Lds[buf][1][0][0];
        const unsigned short* W2b = &Lds[buf][NTILE - 1][0][0];
#pragma unroll
        for (int kc = 0; kc < 2; ++kc) {           // two 32-wide WMMA K-chunks
            Frag a[TM];
#pragma unroll
            for (int i = 0; i < TM; ++i) {
                const unsigned short* p = Ab + (wmL + i * 16 + r16) * LDK + kc * 32 + aoff;
                a[i].q[0] = *(const u32x4*)p;
                a[i].q[1] = *(const u32x4*)(p + 16);
            }
#pragma unroll
            for (int j = 0; j < TN; ++j) {
                Frag b;
                const unsigned short* p = Wb + (wnL + j * 16 + r16) * LDK + kc * 32 + aoff;
                b.q[0] = *(const u32x4*)p;
                b.q[1] = *(const u32x4*)(p + 16);
#pragma unroll
                for (int i = 0; i < TM; ++i) acc[i][j] = wmma_bf16(a[i], b, acc[i][j]);
                if constexpr (EPI == 4) {
                    Frag b2;
                    const unsigned short* p2 =
                        W2b + (wnL + j * 16 + r16) * LDK + kc * 32 + aoff;
                    b2.q[0] = *(const u32x4*)p2;
                    b2.q[1] = *(const u32x4*)(p2 + 16);
#pragma unroll
                    for (int i = 0; i < TM; ++i)
                        acc2[i][j] = wmma_bf16(a[i], b2, acc2[i][j]);
                }
            }
        }
        buf ^= 1;
    }
    // Epilogue: C layout — VGPR r holds (m0+r+mhi, n0+lane&15)
#pragma unroll
    for (int i = 0; i < TM; ++i)
#pragma unroll
        for (int j = 0; j < TN; ++j)
#pragma unroll
            for (int r = 0; r < 8; ++r) {
                const int mm = wm + i * 16 + r + mhi;
                const int nn = wn + j * 16 + r16;
                const size_t idx = (size_t)mm * N + nn;
                float cv = acc[i][j][r];
                if constexpr (EPI == 0) {
                    ((unsigned short*)out)[idx] = f32_bf16(cv);
                } else if constexpr (EPI == 1) {
                    ((unsigned short*)out)[idx] = f32_bf16(cv + bias[nn]);
                } else if constexpr (EPI == 2) {
                    ((float*)out)[idx] = cv + bias[nn] + resid[idx];
                } else if constexpr (EPI == 3) {
                    ((float*)out)[idx] = cv + resid[idx];
                } else {
                    float gg = cv, vv = acc2[i][j][r];
                    float si = gg / (1.0f + __expf(-gg));
                    ((unsigned short*)out)[idx] = f32_bf16(si * vv);
                }
            }
}

// ---------------------------------------------------------------------------
// Spatial-gating bias: bias[b,h,q,k] = scale * sum_d qs[b,q+1,h,d]*ks[b,k+1,h,d]
// One wave per (b,h); K=16 zero-padded to 32 via branchless masks; 16 WMMAs.
// ---------------------------------------------------------------------------
__global__ __launch_bounds__(32)
void sgbias_kernel(const unsigned short* __restrict__ qs,
                   const unsigned short* __restrict__ ksb,
                   const float* __restrict__ scale_p,
                   float* __restrict__ biasb) {
    const int bh = blockIdx.x;
    const int b = bh >> 4, h = bh & 15;
    const int lane = threadIdx.x & 31;
    const int r16  = lane & 15;
    const int mhi  = (lane < 16) ? 0 : 8;
    const unsigned msk = (lane < 16) ? 0xFFFFFFFFu : 0u;   // unused; pad is K 16..31 in q[1] of both halves
    const float scale = scale_p[0];

    f32x8 acc[4][4] = {};
    Frag a[4];
#pragma unroll
    for (int i = 0; i < 4; ++i) {
        const int srow = b * SS + 1 + i * 16 + r16;
        const unsigned short* p =
            qs + (size_t)srow * (HH * DSS) + h * DSS + ((lane < 16) ? 0 : 8);
        a[i].q[0] = *(const u32x4*)p;   // K 0..7 (lo lanes) / 8..15 (hi lanes)
        a[i].q[1] = a[i].q[0] & 0u;     // K 16..31 zero pad
    }
#pragma unroll
    for (int j = 0; j < 4; ++j) {
        Frag bf;
        const int srow = b * SS + 1 + j * 16 + r16;
        const unsigned short* p = ksb + (size_t)srow * (HH * DSS) + h * DSS + ((lane < 16) ? 0 : 8);
        bf.q[0] = *(const u32x4*)p;              // K 0..7 (lo lanes) / 8..15 (hi lanes)
        bf.q[1] = bf.q[0] & 0u;                  // K 16..31 zero pad
#pragma unroll
        for (int i = 0; i < 4; ++i) acc[i][j] = wmma_bf16(a[i], bf, acc[i][j]);
    }
    float* ob = biasb + (size_t)bh * 64 * 64;
#pragma unroll
    for (int i = 0; i < 4; ++i)
#pragma unroll
        for (int j = 0; j < 4; ++j)
#pragma unroll
            for (int r = 0; r < 8; ++r) {
                const int q  = i * 16 + r + mhi;
                const int kk = j * 16 + r16;
                ob[(size_t)q * 64 + kk] = acc[i][j][r] * scale;
            }
}

// ---------------------------------------------------------------------------
// Attention per (b,h): scores = qk^T/8 + bias (padded), softmax, ctx = P@V.
// 128 threads (4 waves); K/V^T staged in LDS, S=65 padded to 80 (P-K to 96).
// ---------------------------------------------------------------------------
__global__ __launch_bounds__(128)
void attention_kernel(const unsigned short* __restrict__ qkv,
                      const float* __restrict__ biasb,
                      unsigned short* __restrict__ ctx) {
    const int bh = blockIdx.x;
    const int b = bh >> 4, h = bh & 15;
    const int tid  = threadIdx.x;
    const int lane = tid & 31;
    const int wave = tid >> 5;
    const int r16  = lane & 15;
    const int aoff = (lane < 16) ? 0 : 8;
    const int boff = (lane < 16) ? 0 : 16;
    const int mhi  = (lane < 16) ? 0 : 8;

    __shared__ alignas(16) unsigned short Ks[80][64];   // K rows, zero-padded
    __shared__ alignas(16) unsigned short Vt[64][96];   // V transposed, padded
    __shared__ float                      scb[4][16][80];
    __shared__ alignas(16) unsigned short pb[4][16][96];

    const size_t base = (size_t)b * SS * (3 * DD) + (size_t)h * DHH;
    for (int idx = tid; idx < 80 * 64; idx += 128) {
        const int r = idx >> 6, d = idx & 63;
        Ks[r][d] = (r < SS) ? qkv[base + (size_t)r * (3 * DD) + DD + d] : 0;
    }
    for (int idx = tid; idx < 64 * 96; idx += 128) {
        const int d = idx / 96, r = idx % 96;
        Vt[d][r] = (r < SS) ? qkv[base + (size_t)r * (3 * DD) + 2 * DD + d] : 0;
    }
    __syncthreads();

    const float* bb = biasb + (size_t)bh * 64 * 64;

    for (int t = wave; t < 5; t += 4) {          // q tiles of 16 rows
        Frag aq[2];
        {
            int row = t * 16 + r16; if (row > SS - 1) row = SS - 1;  // clamp pad rows
            const unsigned short* p = qkv + base + (size_t)row * (3 * DD);
#pragma unroll
            for (int kc = 0; kc < 2; ++kc) {
                aq[kc].q[0] = *(const u32x4*)(p + kc * 32 + aoff);
                aq[kc].q[1] = *(const u32x4*)(p + kc * 32 + aoff + 16);
            }
        }
#pragma unroll
        for (int j = 0; j < 5; ++j) {            // k tiles
            f32x8 acc = {};
#pragma unroll
            for (int kc = 0; kc < 2; ++kc) {
                Frag bk;
                const unsigned short* p = &Ks[j * 16 + r16][kc * 32 + aoff];
                bk.q[0] = *(const u32x4*)p;
                bk.q[1] = *(const u32x4*)(p + 16);
                acc = wmma_bf16(aq[kc], bk, acc);
            }
#pragma unroll
            for (int r = 0; r < 8; ++r) {
                const int qr = r + mhi;
                const int qg = t * 16 + qr;
                const int kg = j * 16 + r16;
                float s = acc[r] * 0.125f;       // 1/sqrt(DH)
                if (qg >= 1 && qg < SS && kg >= 1 && kg < SS)
                    s += bb[(size_t)(qg - 1) * 64 + (kg - 1)];
                if (kg >= SS) s = -1e30f;
                scb[wave][qr][j * 16 + r16] = s;
            }
        }
        // per-row softmax (lanes 0..15 own one row each)
        if (lane < 16) {
            float mx = -1e30f;
            for (int c = 0; c < SS; ++c) mx = fmaxf(mx, scb[wave][lane][c]);
            float sum = 0.f;
            for (int c = 0; c < SS; ++c) sum += __expf(scb[wave][lane][c] - mx);
            const float rinv = 1.0f / sum;
            for (int c = 0; c < 96; ++c) {
                float p = (c < SS) ? __expf(scb[wave][lane][c] - mx) * rinv : 0.f;
                pb[wave][lane][c] = f32_bf16(p);
            }
        }
        // ctx = P(16x96) @ V(96x64)
#pragma unroll
        for (int jn = 0; jn < 4; ++jn) {
            f32x8 acc = {};
#pragma unroll
            for (int kc = 0; kc < 3; ++kc) {
                Frag ap, bv;
                const unsigned short* pa = &pb[wave][r16][kc * 32 + aoff];
                ap.q[0] = *(const u32x4*)pa;
                ap.q[1] = *(const u32x4*)(pa + 16);
                const unsigned short* pv = &Vt[jn * 16 + r16][kc * 32 + aoff];
                bv.q[0] = *(const u32x4*)pv;
                bv.q[1] = *(const u32x4*)(pv + 16);
                acc = wmma_bf16(ap, bv, acc);
            }
#pragma unroll
            for (int r = 0; r < 8; ++r) {
                const int qg = t * 16 + r + mhi;
                if (qg < SS)
                    ctx[((size_t)b * SS + qg) * DD + h * DHH + jn * 16 + r16] =
                        f32_bf16(acc[r]);
            }
        }
    }
}

// ---------------------------------------------------------------------------
// Host orchestration
// ---------------------------------------------------------------------------
extern "C" void kernel_launch(void* const* d_in, const int* in_sizes, int n_in,
                              void* d_out, int out_size, void* d_ws, size_t ws_size,
                              hipStream_t stream) {
    const float* src      = (const float*)d_in[0];
    const float* g1       = (const float*)d_in[1];
    const float* b1       = (const float*)d_in[2];
    const float* in_w     = (const float*)d_in[3];
    const float* in_b     = (const float*)d_in[4];
    const float* out_w    = (const float*)d_in[5];
    const float* out_b    = (const float*)d_in[6];
    const float* sg_qw    = (const float*)d_in[7];
    const float* sg_kw    = (const float*)d_in[8];
    const float* sg_scale = (const float*)d_in[9];
    const float* g2       = (const float*)d_in[10];
    const float* b2       = (const float*)d_in[11];
    const float* gv_w     = (const float*)d_in[12];
    const float* w3_w     = (const float*)d_in[13];

    size_t off = 0;
    auto alloc = [&](size_t bytes) -> void* {
        off = (off + 255) & ~(size_t)255;
        void* p = (char*)d_ws + off;
        off += bytes;
        return p;
    };
    const size_t NT = NTOK;
    unsigned short* src_bf = (unsigned short*)alloc(NT * DD * 2);
    unsigned short* n1buf  = (unsigned short*)alloc(NT * DD * 2);
    unsigned short* qkvbuf = (unsigned short*)alloc(NT * 3 * DD * 2);
    unsigned short* qsbuf  = (unsigned short*)alloc(NT * 256 * 2);
    unsigned short* ksbuf  = (unsigned short*)alloc(NT * 256 * 2);
    unsigned short* ctxbuf = (unsigned short*)alloc(NT * DD * 2);
    unsigned short* n2buf  = (unsigned short*)alloc(NT * DD * 2);
    unsigned short* hbuf   = (unsigned short*)alloc(NT * DFFD * 2);
    unsigned short* w_in   = (unsigned short*)alloc((size_t)3 * DD * DD * 2);
    unsigned short* w_out  = (unsigned short*)alloc((size_t)DD * DD * 2);
    unsigned short* w_sgq  = (unsigned short*)alloc((size_t)256 * DD * 2);
    unsigned short* w_sgk  = (unsigned short*)alloc((size_t)256 * DD * 2);
    unsigned short* w_gv   = (unsigned short*)alloc((size_t)2 * DFFD * DD * 2);
    unsigned short* w_w3   = (unsigned short*)alloc((size_t)DD * DFFD * 2);
    float* xres  = (float*)alloc(NT * DD * 4);
    float* biasb = (float*)alloc((size_t)BB * HH * 64 * 64 * 4);

    auto castw = (void(*)(const float*, unsigned short*, size_t, hipStream_t)) nullptr;
    (void)castw;
    auto launch_cast = [&](const float* s, unsigned short* d, size_t n) {
        int n4 = (int)(n / 4);
        cast_bf16_kernel<<<(n4 + 255) / 256, 256, 0, stream>>>(s, d, n4);
    };
    launch_cast(in_w,  w_in,  (size_t)3 * DD * DD);
    launch_cast(out_w, w_out, (size_t)DD * DD);
    launch_cast(sg_qw, w_sgq, (size_t)256 * DD);
    launch_cast(sg_kw, w_sgk, (size_t)256 * DD);
    launch_cast(gv_w,  w_gv,  (size_t)2 * DFFD * DD);
    launch_cast(w3_w,  w_w3,  (size_t)DD * DFFD);

    // LN1 + cast src to bf16
    ln_cast_kernel<<<NT, 256, 0, stream>>>(src, g1, b1, n1buf, src_bf);

    // spatial gating projections (run over all rows; row 0 unused downstream)
    gemm_wmma_kernel<128, 128, 0><<<dim3(256 / 128, NT / 128), 128, 0, stream>>>(
        src_bf, w_sgq, nullptr, nullptr, qsbuf, (int)NT, 256, DD);
    gemm_wmma_kernel<128, 128, 0><<<dim3(256 / 128, NT / 128), 128, 0, stream>>>(
        src_bf, w_sgk, nullptr, nullptr, ksbuf, (int)NT, 256, DD);
    sgbias_kernel<<<BB * HH, 32, 0, stream>>>(qsbuf, ksbuf, sg_scale, biasb);

    // QKV projection (+in_b) -> bf16
    gemm_wmma_kernel<128, 128, 1><<<dim3(3 * DD / 128, NT / 128), 128, 0, stream>>>(
        n1buf, w_in, in_b, nullptr, qkvbuf, (int)NT, 3 * DD, DD);

    // attention
    attention_kernel<<<BB * HH, 128, 0, stream>>>(qkvbuf, biasb, ctxbuf);

    // out projection + out_b + src residual -> x (f32)
    gemm_wmma_kernel<128, 128, 2><<<dim3(DD / 128, NT / 128), 128, 0, stream>>>(
        ctxbuf, w_out, out_b, src, xres, (int)NT, DD, DD);

    // LN2
    ln_cast_kernel<<<NT, 256, 0, stream>>>(xres, g2, b2, n2buf, nullptr);

    // fused SwiGLU GEMM: h = silu(n2@Wg^T) * (n2@Wv^T)
    gemm_wmma_kernel<64, 64, 4><<<dim3(DFFD / 64, NT / 64), 128, 0, stream>>>(
        n2buf, w_gv, nullptr, nullptr, hbuf, (int)NT, DFFD, DD);

    // final: out = x + h @ w3^T  (f32 to d_out)
    gemm_wmma_kernel<128, 128, 3><<<dim3(DD / 128, NT / 128), 128, 0, stream>>>(
        hbuf, w_w3, nullptr, xres, d_out, (int)NT, DD, DFFD);

    (void)in_sizes; (void)n_in; (void)out_size; (void)ws_size;
}
